// SMorphLayer_MAE_20873541059122
// MI455X (gfx1250) — matrix operand order repaired and checked by reading, not compiled, not run on hardware
//
#include <hip/hip_runtime.h>

// ---------------------------------------------------------------------------
// SMorph layer (ALPHA=1) as WMMA GEMM on gfx1250 (MI455X), v4.
//
// exp(x+k) = exp(x)*exp(k)  (alpha=1)  =>
//   D[f]   = sum_p e^{x_p} E_pf                       (E  = e^{k})
//   Num[f] = sum_p (x_p e^{x_p}) E_pf + e^{x_p} KE_pf (KE = k e^{k})
//   out    = Num1/D1 + Num2/D2 + bias   (branch2: x -> -x, k -> k2)
//
// v4 vs v3:
//  * Gathers use GVS addressing: uniform SGPR base (pk table) + 32-bit VGPR
//    byte offset -> global_load_b64 saddr form, 1 v_add_u32 per gather
//    (was mov+ashr+add_u64 per gather).
//  * Epilogue uses v_rcp_f32 + FMA instead of 16 full f32 divides.
//
// Hot loop per k-step/lane: 2 ds_load_b128 (offsets) + 8 global_load_b64
// (packed {u1,v1|u2,v2} f16 quads, precomputed once per x element) +
// 8 ds_load_b128 (B fragments) + 4 v_wmma_f32_16x16x32_f16. No exp/cvt/mul.
// ---------------------------------------------------------------------------

typedef __attribute__((ext_vector_type(16))) _Float16 v16h;
typedef __attribute__((ext_vector_type(8)))  float    v8f;
typedef unsigned long long u64;

#define NB      8
#define CIN     4
#define HIN     64
#define WIN     64
#define KS      5
#define FOUT    16
#define HO      60
#define WO      60
#define PDIM    100            // KS*KS*CIN, patch length, order (kh, kw, c)
#define PPAD    112            // offset-table pad (max p touched = 111)
#define KSTEPS  7              // 224 / 32
#define NPIX    (NB * HO * WO) // 28800
#define NTILES  (NPIX / 16)    // 1800
#define NXELEM  (NB * CIN * HIN * WIN)              // 131072 x elements
#define FRAG_HALFS 512         // 32 lanes * 16 halves per B fragment
#define NFRAG_HALFS (2 * KSTEPS * 2 * FRAG_HALFS)   // 14336 halves = 28 KB
#define PK_WS_OFFSET 32768     // pk table placed after B fragments in ws

// ---------------------------------------------------------------------------
// Precompute kernel: per x element pack {u1,v1,u2,v2} as 4 f16 in one u64.
// ---------------------------------------------------------------------------
__global__ void smorph_precompute(const float* __restrict__ x,
                                  u64* __restrict__ pk) {
    int i = blockIdx.x * blockDim.x + threadIdx.x;
    if (i >= NXELEM) return;
    const float xv = x[i];
    const float ep = __expf(xv);
    const float em = __expf(-xv);
    union { _Float16 h[4]; u64 q; } u;
    u.h[0] = (_Float16)ep;          // u1  (A slot 2p, low half of dword)
    u.h[1] = (_Float16)(xv * ep);   // v1  (A slot 2p+1)
    u.h[2] = (_Float16)em;          // u2
    u.h[3] = (_Float16)(-xv * em);  // v2
    pk[i] = u.q;
}

// ---------------------------------------------------------------------------
// Setup kernel: bake B fragments (f16) into workspace, interleaved K-order.
// B-fragment element (16-bit B, 32x16): col f = lane&15,
//   row kk = 32*s + (j&7) + 2*(j&8) + 8*(lane>=16);  p = kk>>1, isV = kk&1.
//   nsel=0 (denominator): B = isV ? 0     : e^{k[p,f]}
//   nsel=1 (numerator):   B = isV ? e^{k} : k*e^{k}
// ---------------------------------------------------------------------------
__global__ void smorph_build_bfrags(const float* __restrict__ k1,
                                    const float* __restrict__ k2,
                                    _Float16* __restrict__ bfr) {
    int idx = blockIdx.x * blockDim.x + threadIdx.x;
    if (idx >= NFRAG_HALFS) return;

    const int j    = idx & 15;
    const int lane = (idx >> 4) & 31;
    const int nsel = (idx >> 9) & 1;
    const int s    = (idx >> 10) % KSTEPS;
    const int br   = idx / (KSTEPS * 2 * FRAG_HALFS);

    const float* km = br ? k2 : k1;                  // flattened (P, F)
    const int kk  = 32 * s + (j & 7) + ((j & 8) << 1) + ((lane >= 16) ? 8 : 0);
    const int p   = kk >> 1;
    const int isV = kk & 1;
    const int col = lane & 15;

    float v = 0.0f;
    if (p < PDIM) {
        const float kv = km[p * FOUT + col];
        const float ek = __expf(kv);
        v = (nsel == 0) ? (isV ? 0.0f : ek)
                        : (isV ? ek   : kv * ek);
    }
    bfr[idx] = (_Float16)v;
}

// ---------------------------------------------------------------------------
// Main kernel: 128 threads = 4 waves; one wave per 16x16 output tile.
// ---------------------------------------------------------------------------
__global__ void __launch_bounds__(128)
smorph_wmma_kernel(const u64* __restrict__ pk,
                   const _Float16* __restrict__ bfr,
                   const float* __restrict__ bias,
                   float* __restrict__ out) {
    __shared__ __align__(16) int      poff[PPAD];       // patch p -> BYTE offset into pk
    __shared__ __align__(16) _Float16 bsh[NFRAG_HALFS]; // 28 KB B fragments

    const int tid = threadIdx.x;

    // ---- cooperative setup: pre-scaled offset table + B fragments -> LDS ----
    if (tid < PPAD) {
        int off = 0;
        if (tid < PDIM) {
            const int kh  = tid / (KS * CIN);           // /20
            const int rem = tid - kh * (KS * CIN);
            const int kw  = rem >> 2;                   // /CIN
            const int c   = rem & 3;                    // %CIN
            off = (c * (HIN * WIN) + kh * WIN + kw) * 8; // byte offset (u64 elems)
        }
        poff[tid] = off;                                // pad -> 0 (in-bounds dummy)
    }
    {
        const uint4* src = (const uint4*)bfr;           // 28672 B = 1792 x 16 B
        uint4*       dst = (uint4*)bsh;
        #pragma unroll
        for (int i = 0; i < 14; ++i)
            dst[tid + 128 * i] = src[tid + 128 * i];
    }
    __syncthreads();

    const int lane = tid & 31;
    const int wave = tid >> 5;
    const int tile = blockIdx.x * 4 + wave;             // exact grid: tile < NTILES

    // ---- this lane's matrix row = one output pixel ----
    const int row   = lane & 15;                        // M
    const int pixel = tile * 16 + row;
    const int n  = pixel / (HO * WO);
    const int r  = pixel - n * (HO * WO);
    const int ho = r / WO;
    const int wo = r - ho * WO;
    // 32-bit per-lane byte offset of this pixel's patch origin in the pk table.
    const unsigned rowbytes =
        (unsigned)((n * (CIN * HIN * WIN) + ho * WIN + wo) * 8);
    const char* __restrict__ pkb = (const char*)pk;     // UNIFORM base -> SADDR form
    const int pbase = (lane >= 16) ? 4 : 0;             // A-fragment K shift /2

    v8f acc_d1 = {}, acc_n1 = {}, acc_d2 = {}, acc_n2 = {};
    const v16h* __restrict__ bls = (const v16h*)bsh;

    #pragma unroll
    for (int s = 0; s < KSTEPS; ++s) {
        // A fragments: one b64 gather per patch element supplies one packed
        // (u,v) dword per branch. GVS addressing: sgpr base + u32 voffset.
        union { v16h v; unsigned int w[8]; } A1, A2;
        #pragma unroll
        for (int half = 0; half < 2; ++half) {
            const int  p0   = 16 * s + pbase + 8 * half;
            const int4 offs = *(const int4*)&poff[p0];  // ds_load_b128
            #pragma unroll
            for (int t = 0; t < 4; ++t) {
                const int      p    = p0 + t;
                const unsigned voff = rowbytes + (unsigned)((&offs.x)[t]);
                u64 w = *(const u64*)(pkb + voff);      // global_load_b64 (saddr)
                if (p >= PDIM) w = 0ull;                // materializes only at s==6
                A1.w[4 * half + t] = (unsigned int)w;         // {u1,v1}
                A2.w[4 * half + t] = (unsigned int)(w >> 32); // {u2,v2}
            }
        }

        // B fragments from LDS (ds_load_b128 x2 per v16h, immediate offsets).
        const v16h b1d = bls[((0 * KSTEPS + s) * 2 + 0) * 32 + lane];
        const v16h b1n = bls[((0 * KSTEPS + s) * 2 + 1) * 32 + lane];
        const v16h b2d = bls[((1 * KSTEPS + s) * 2 + 0) * 32 + lane];
        const v16h b2n = bls[((1 * KSTEPS + s) * 2 + 1) * 32 + lane];

        acc_d1 = __builtin_amdgcn_wmma_f32_16x16x32_f16(false, A1.v, false, b1d, (short)0, acc_d1, false, false);
        acc_n1 = __builtin_amdgcn_wmma_f32_16x16x32_f16(false, A1.v, false, b1n, (short)0, acc_n1, false, false);
        acc_d2 = __builtin_amdgcn_wmma_f32_16x16x32_f16(false, A2.v, false, b2d, (short)0, acc_d2, false, false);
        acc_n2 = __builtin_amdgcn_wmma_f32_16x16x32_f16(false, A2.v, false, b2n, (short)0, acc_n2, false, false);
    }

    // ---- Epilogue: out = Num1/D1 + Num2/D2 + bias, via v_rcp_f32 + FMA ----
    // C/D layout: col f = lane&15; element i -> M = i + 8*(lane>=16).
    const int   f    = lane & 15;
    const float bv   = bias[f];
    const int   mofs = (lane >= 16) ? 8 : 0;
    #pragma unroll
    for (int i = 0; i < 8; ++i) {
        const int px = tile * 16 + i + mofs;
        const int n2 = px / (HO * WO);
        const int r2 = px - n2 * (HO * WO);
        const int h2 = r2 / WO;
        const int w2 = r2 - h2 * WO;
        const float r1 = __builtin_amdgcn_rcpf(acc_d1[i]);   // v_rcp_f32
        const float r2f = __builtin_amdgcn_rcpf(acc_d2[i]);  // v_rcp_f32
        const float val = fmaf(acc_n1[i], r1, fmaf(acc_n2[i], r2f, bv));
        out[((n2 * FOUT + f) * HO + h2) * WO + w2] = val;
    }
}

// ---------------------------------------------------------------------------
extern "C" void kernel_launch(void* const* d_in, const int* in_sizes, int n_in,
                              void* d_out, int out_size, void* d_ws, size_t ws_size,
                              hipStream_t stream) {
    (void)in_sizes; (void)n_in; (void)out_size; (void)ws_size;

    const float* x    = (const float*)d_in[0];   // (8,4,64,64)
    const float* k1   = (const float*)d_in[1];   // (5,5,4,16)
    const float* k2   = (const float*)d_in[2];   // (5,5,4,16)
    const float* bias = (const float*)d_in[3];   // (16,)
    float*       out  = (float*)d_out;           // (8,16,60,60) f32

    _Float16* bfr = (_Float16*)d_ws;                          // 28 KB B fragments
    u64*      pkt = (u64*)((char*)d_ws + PK_WS_OFFSET);       // 1 MB packed exp table

    smorph_precompute<<<(NXELEM + 255) / 256, 256, 0, stream>>>(x, pkt);
    smorph_build_bfrags<<<(NFRAG_HALFS + 255) / 256, 256, 0, stream>>>(k1, k2, bfr);

    // 1800 tiles, 4 waves/block -> 450 blocks exactly (no remainder waves).
    smorph_wmma_kernel<<<NTILES / 4, 128, 0, stream>>>(pkt, bfr, bias, out);
}
